// DynamicFNO3d_51376398795167
// MI455X (gfx1250) — compile-verified, matrix operand order checked by reading, not run on hardware
//
#include <hip/hip_runtime.h>
#include <hip/hip_bf16.h>

typedef __attribute__((ext_vector_type(16))) _Float16 v16h;
typedef __attribute__((ext_vector_type(8)))  float    v8f;

#define PI2F 6.28318530717958647692f

// f16 A-matrix 16x32 lane layout: lane L holds row (L&15); K set depends on half = L>>4.
__device__ __forceinline__ int kmap(int half, int e) {
  return (e < 8) ? (half * 8 + e) : (16 + half * 8 + (e - 8));
}
// Branchless exact-ish GELU (tanh form; no erff branches).
__device__ __forceinline__ float gelu_f(float v) {
  float u = 0.7978845608f * (v + 0.044715f * v * v * v);
  float e = __expf(-2.f * fabsf(u));      // in (0,1], no overflow path
  float t = __fdividef(1.f - e, 1.f + e); // tanh(|u|)
  t = copysignf(t, u);
  return 0.5f * v * (1.f + t);
}

// ---------------------------------------------------------------------------
// Twiddle bases (f16): BZF[64x16] | BYF[128x32] | BXI[32x128] | W1H[8][32][16]
// plus f32 trig table [2x368] for the Z-inverse.
// ---------------------------------------------------------------------------
__global__ void init_basis(_Float16* bas, float* trig, const float* __restrict__ w1) {
  for (int i = threadIdx.x; i < 1024 + 4096 + 4096; i += blockDim.x) {
    float v = 0.f;
    if (i < 1024) {                       // Z-forward: 46 samples -> 8 complex modes
      int k = i >> 4, n = i & 15;
      if (k < 46) {
        int m = n & 7;
        float th = PI2F * (float)((m * k) % 46) / 46.f;
        v = (n < 8) ? cosf(th) : -sinf(th);
      }
      bas[i] = (_Float16)v;
    } else if (i < 5120) {                // Y/X-forward: 64 complex -> 16 complex modes
      int j = i - 1024;
      int k = j >> 5, n = j & 31;
      int m = n & 15; int km = (m < 8) ? m : 48 + m;
      int y = (k < 64) ? k : k - 64;
      float th = PI2F * (float)((km * y) & 63) / 64.f;
      bool isIm = (n >= 16);
      if (k < 64) v = isIm ? -sinf(th) : cosf(th);   // real input rows
      else        v = isIm ?  cosf(th) : sinf(th);   // imag input rows
      bas[i] = (_Float16)v;
    } else {                              // X/Y-inverse: 16 complex modes -> 64 complex, 1/64
      int j = i - 5120;
      int k = j >> 7, n = j & 127;
      int x = n & 63; bool isImOut = (n >= 64);
      int m = k & 15; bool isImIn  = (k >= 16);
      int km = (m < 8) ? m : 48 + m;
      float th = PI2F * (float)((km * x) & 63) / 64.f;
      if (!isImIn) v = isImOut ?  sinf(th) : cosf(th);
      else         v = isImOut ?  cosf(th) : -sinf(th);
      bas[i] = (_Float16)(v * (1.f / 64.f));
    }
  }
  // fc1_w pre-packed in B-fragment order: [jtile][k(32)][n(16)], zero-padded k>=20
  _Float16* w1h = bas + 9216;
  for (int i = threadIdx.x; i < 4096; i += blockDim.x) {
    int j = i >> 9, k = (i >> 4) & 31, n = i & 15;
    float v = (k < 20) ? w1[(j * 16 + n) * 20 + k] : 0.f;
    w1h[i] = (_Float16)v;
  }
  for (int i = threadIdx.x; i < 368; i += blockDim.x) { // Z-inverse trig table
    int k = i / 46, z = i % 46;
    float th = PI2F * (float)((k * z) % 46) / 46.f;
    trig[i]       = cosf(th);
    trig[i + 368] = sinf(th);
  }
}

// ---------------------------------------------------------------------------
// fc0: concat grids, 13->20 per point; pad region z>=40 zeroed.
// h layout: [b][c][x][y][z], z fastest.  1,507,328 threads.
// ---------------------------------------------------------------------------
__global__ __launch_bounds__(256) void fc0_kernel(const float* __restrict__ xin,
                                                  const float* __restrict__ w,
                                                  const float* __restrict__ bvec,
                                                  float* __restrict__ h) {
  int tid = blockIdx.x * 256 + threadIdx.x;
  int z = tid % 46; int p = tid / 46;
  int y = p & 63; p >>= 6;
  int x = p & 63; int b = p >> 6;
  if (z >= 40) {
    for (int c = 0; c < 20; ++c)
      h[((b * 20 + c) * 4096 + x * 64 + y) * 46 + z] = 0.f;
    return;
  }
  float in[13];
  const float* xp = xin + (((b * 64 + x) * 64 + y) * 40 + z) * 10;
  #pragma unroll
  for (int i = 0; i < 10; ++i) in[i] = xp[i];
  in[10] = x * (1.f / 63.f); in[11] = y * (1.f / 63.f); in[12] = z * (1.f / 39.f);
  for (int c = 0; c < 20; ++c) {
    float acc = bvec[c];
    #pragma unroll
    for (int i = 0; i < 13; ++i) acc += in[i] * w[c * 13 + i];
    h[((b * 20 + c) * 4096 + x * 64 + y) * 46 + z] = acc;
  }
}

// ---------------------------------------------------------------------------
// Generic WMMA DFT-GEMM.  One wave per block, one 16x16 output tile per (tm,tn).
// A-tile staged in LDS as f16 [16][K] (row padded +8 to break bank conflicts).
// STG 0: Z-fwd   h(rows=bcxy,K=46->64)        -> Hz  [bcx][kz][y][reim]
// STG 1: Y-fwd   Hz(rows=bcx*8+kz,K=128)      -> Hyz [bc][ym][kz][x][reim]
// STG 2: X-fwd   Hyz(rows=bc*128+ym*8+kz)     -> Hc  [bc][xm][ym][kz][reim]
// STG 3: X-inv   Gc(rows=bc*128+ym*8+kz,K=32) -> Gx  [bc][x][ym][kz][reim]
// STG 4: Y-inv   Gx(rows=bc*512+x*8+kz,K=32)  -> Gy  [bc][x][y][kz][reim]
// ---------------------------------------------------------------------------
template <int STG>
__global__ __launch_bounds__(32) void dft_gemm(const float* __restrict__ A,
                                               const _Float16* __restrict__ basis,
                                               float* __restrict__ D) {
  constexpr int K  = (STG == 0) ? 64 : (STG <= 2) ? 128 : 32;
  constexpr int NC = (STG == 0) ? 16 : (STG <= 2) ? 32  : 128;
  constexpr int KP = K + 8;
  __shared__ _Float16 lds16[16 * KP];

  const int lane = threadIdx.x;
  const int half = lane >> 4, lc = lane & 15;
  const int tm = blockIdx.x, tn = blockIdx.y;
  const int colB = tn * 16 + lc;

  // ---- cooperative, coalesced tile load -> LDS (f16, re-laid out as [row][k])
  if constexpr (STG == 0) {
    const float* base = A + tm * 736;
    for (int i = lane; i < 1024; i += 32) {         // [row][k64]
      int row = i >> 6, k = i & 63;
      int kk = (k < 46) ? k : 45;
      float v = base[row * 46 + kk];
      lds16[row * KP + k] = (_Float16)((k < 46) ? v : 0.f);
    }
  } else if constexpr (STG == 1 || STG == 2) {
    const float* base = A + tm * 2048;              // contiguous 2048 floats
    for (int i = lane; i < 2048; i += 32) {
      int row = i >> 7;                             // (bcxL*8+kz) or ymkzL
      int k = (i & 1) * 64 + ((i >> 1) & 63);       // reim*64 + coord
      lds16[row * KP + k] = (_Float16)base[i];
    }
  } else if constexpr (STG == 3) {
    int bc = tm >> 3;
    const float* base = A + bc * 4096 + (tm & 7) * 32;   // 16 segments of 32
    for (int i = lane; i < 512; i += 32) {
      int xm = i >> 5, j = i & 31;                  // j = ymL*16 + kz*2 + reim
      int row = ((j >> 4) & 1) * 8 + ((j >> 1) & 7);
      int k = (j & 1) * 16 + xm;
      lds16[row * KP + k] = (_Float16)base[xm * 256 + j];
    }
  } else {
    const float* base = A + tm * 512;               // contiguous 512 floats
    for (int i = lane; i < 512; i += 32) {
      int off = i & 255;                            // ym*16 + kz*2 + reim
      int row = (i >> 8) * 8 + ((off >> 1) & 7);
      int k = (off & 1) * 16 + (off >> 4);
      lds16[row * KP + k] = (_Float16)base[i];
    }
  }
  __syncthreads();

  // ---- WMMA accumulate
  v8f c = {0.f, 0.f, 0.f, 0.f, 0.f, 0.f, 0.f, 0.f};
  for (int kc = 0; kc < K; kc += 32) {
    v16h a, bf;
    #pragma unroll
    for (int e = 0; e < 16; ++e) {
      int k = kc + kmap(half, e);
      a[e]  = lds16[lc * KP + k];
      bf[e] = basis[k * NC + colB];
    }
    c = __builtin_amdgcn_wmma_f32_16x16x32_f16(false, a, false, bf,
                                               (short)0, c, false, false);
  }

  // ---- scatter store into next stage's layout
  #pragma unroll
  for (int r = 0; r < 8; ++r) {
    int row = tm * 16 + half * 8 + r;
    int n = colB;
    float v = c[r];
    if constexpr (STG == 0) {
      int bcx = row >> 6, y = row & 63;
      D[bcx * 1024 + (n & 7) * 128 + y * 2 + (n >> 3)] = v;
    } else if constexpr (STG == 1) {
      int bc = row >> 9, x = (row >> 3) & 63, kz = row & 7;
      D[bc * 16384 + ((n & 15) * 8 + kz) * 128 + x * 2 + (n >> 4)] = v;
    } else if constexpr (STG == 2) {
      int bc = row >> 7, ym = (row >> 3) & 15, kz = row & 7;
      D[bc * 4096 + (n & 15) * 256 + ym * 16 + kz * 2 + (n >> 4)] = v;
    } else if constexpr (STG == 3) {
      int bc = row >> 7, ym = (row >> 3) & 15, kz = row & 7;
      D[bc * 16384 + (n & 63) * 256 + ym * 16 + kz * 2 + (n >> 6)] = v;
    } else {
      int bc = row >> 9, x = (row >> 3) & 63, kz = row & 7;
      D[bc * 65536 + x * 1024 + (n & 63) * 16 + kz * 2 + (n >> 6)] = v;
    }
  }
}

// ---------------------------------------------------------------------------
// Gate: mean |Hc| over (c, xm<8, ym<8) per (b,kz), then sigmoid(avg @ gw.T + gb)
// ---------------------------------------------------------------------------
__global__ __launch_bounds__(256) void gate_kernel(const float* __restrict__ Hc,
                                                   const float* __restrict__ gw,
                                                   const float* __restrict__ gb,
                                                   float* __restrict__ gate, int layer) {
  __shared__ float red[256];
  __shared__ float favg[8];
  int b = blockIdx.x, t = threadIdx.x;
  int kz = t & 7, slice = t >> 3;
  float s = 0.f;
  for (int idx = slice; idx < 1280; idx += 32) {
    int c = idx >> 6, xy = idx & 63;
    int a = (b * 20 + c) * 4096 + (xy >> 3) * 256 + (xy & 7) * 16 + kz * 2;
    float re = Hc[a], im = Hc[a + 1];
    s += sqrtf(re * re + im * im);
  }
  red[t] = s; __syncthreads();
  for (int stp = 128; stp >= 8; stp >>= 1) {
    if (t < stp) red[t] += red[t + stp];
    __syncthreads();
  }
  if (t < 8) favg[t] = red[t] * (1.f / 1280.f);
  __syncthreads();
  if (t < 20) {
    float zz = gb[layer * 20 + t];
    #pragma unroll
    for (int k = 0; k < 8; ++k) zz += favg[k] * gw[layer * 160 + t * 8 + k];
    gate[b * 20 + t] = __fdividef(1.f, 1.f + __expf(-zz));
  }
}

// ---------------------------------------------------------------------------
// Spectral mix: out[o] = g*(in@wA) + (1-g)*(in@wB) per corner mode (complex).
// ---------------------------------------------------------------------------
__global__ __launch_bounds__(256) void spectral_kernel(const float* __restrict__ Hc,
                                                       const float* __restrict__ gate,
                                                       const float* __restrict__ wA,
                                                       const float* __restrict__ wB,
                                                       float* __restrict__ Gc, int layer) {
  int tid = blockIdx.x * 256 + threadIdx.x;   // 327,680 total
  int o = tid % 20; int r = tid / 20;
  int kz = r & 7; r >>= 3;
  int my = r & 7; r >>= 3;
  int mx = r & 7; r >>= 3;
  int q = r & 3;  int b = r >> 2;
  int xm = (q & 1) ? 8 + mx : mx;
  int ym = (q & 2) ? 8 + my : my;
  float g = gate[b * 20 + o];
  float aRe = 0.f, aIm = 0.f, bRe = 0.f, bIm = 0.f;
  int mofs = (mx * 64 + my * 8 + kz) * 2;
  int wq = (layer * 4 + q) * 400 + o;         // (((l*4+q)*20+i)*20+o)*1024 + mofs
  for (int i = 0; i < 20; ++i) {
    int ha = (b * 20 + i) * 4096 + xm * 256 + ym * 16 + kz * 2;
    float re = Hc[ha], im = Hc[ha + 1];
    int wi = (wq + i * 20) * 1024 + mofs;
    float wr = wA[wi], wim = wA[wi + 1];
    aRe += re * wr - im * wim; aIm += re * wim + im * wr;
    wr = wB[wi]; wim = wB[wi + 1];
    bRe += re * wr - im * wim; bIm += re * wim + im * wr;
  }
  int oa = (b * 20 + o) * 4096 + xm * 256 + ym * 16 + kz * 2;
  Gc[oa]     = g * aRe + (1.f - g) * bRe;
  Gc[oa + 1] = g * aIm + (1.f - g) * bIm;
}

// ---------------------------------------------------------------------------
// Fused Z-inverse (8 modes -> 46 reals, 1/46) + pointwise conv + add + GELU.
// One block per (b,x,y); stages the 20x46 h column + conv weights in LDS.
// ---------------------------------------------------------------------------
__global__ __launch_bounds__(128) void zi_conv_kernel(const float* __restrict__ Gy,
                                                      float* __restrict__ h,
                                                      const float* __restrict__ cw,
                                                      const float* __restrict__ cb,
                                                      const float* __restrict__ trig,
                                                      int layer) {
  __shared__ float hin[920];
  __shared__ float cz[368], sz[368];
  __shared__ float cwl[400];
  int blk = blockIdx.x;
  int y = blk & 63, x = (blk >> 6) & 63, b = blk >> 12;
  int t = threadIdx.x;
  for (int i = t; i < 920; i += 128) {
    int c = i / 46, z = i % 46;
    hin[i] = h[((b * 20 + c) * 4096 + x * 64 + y) * 46 + z];
  }
  for (int i = t; i < 368; i += 128) {
    cz[i] = trig[i];
    sz[i] = trig[i + 368];
  }
  for (int i = t; i < 400; i += 128) cwl[i] = cw[layer * 400 + i];
  __syncthreads();
  bool doGelu = (layer < 3);
  for (int i = t; i < 920; i += 128) {
    int o = i / 46, z = i % 46;
    float x2 = cb[layer * 20 + o];
    #pragma unroll 4
    for (int c = 0; c < 20; ++c) x2 += hin[c * 46 + z] * cwl[o * 20 + c];
    int gb = ((b * 20 + o) * 4096 + x * 64 + y) * 16;
    float s = Gy[gb];
    #pragma unroll
    for (int k = 1; k < 8; ++k)
      s += 2.f * (Gy[gb + 2 * k] * cz[k * 46 + z] - Gy[gb + 2 * k + 1] * sz[k * 46 + z]);
    float v = s * (1.f / 46.f) + x2;
    if (doGelu) v = gelu_f(v);
    h[((b * 20 + o) * 4096 + x * 64 + y) * 46 + z] = v;
  }
}

// ---------------------------------------------------------------------------
// fc1 (WMMA, 8 col tiles, unrolled) + GELU + fc2 (shuffle reduction).
// One wave = 16 points.  A staged via LDS (coalesced, zero-padded);
// B from pre-packed f16 weights in fragment order (no masks, no cvt).
// ---------------------------------------------------------------------------
__global__ __launch_bounds__(32) void fc12_kernel(const float* __restrict__ h,
                                                  const _Float16* __restrict__ w1h,
                                                  const float* __restrict__ b1,
                                                  const float* __restrict__ w2,
                                                  const float* __restrict__ b2,
                                                  float* __restrict__ out) {
  constexpr int KP = 40;
  __shared__ _Float16 lds16[16 * KP];
  int lane = threadIdx.x;
  int half = lane >> 4, lc = lane & 15;
  int tile = blockIdx.x;
  for (int i = lane; i < 16 * KP; i += 32) lds16[i] = (_Float16)0.f;
  __syncthreads();
  int p = tile * 16 + lc;
  int z = p % 40; p /= 40;
  int y = p & 63; p >>= 6;
  int x = p & 63; int b = p >> 6;
  int baseAddr = (b * 20 * 4096 + x * 64 + y) * 46 + z;
  for (int k = half; k < 20; k += 2)                // lanes 0-15 coalesced per k
    lds16[lc * KP + k] = (_Float16)h[baseAddr + k * 188416];
  __syncthreads();
  v16h a;
  #pragma unroll
  for (int e = 0; e < 16; ++e) a[e] = lds16[lc * KP + kmap(half, e)];
  float acc[8];
  #pragma unroll
  for (int r = 0; r < 8; ++r) acc[r] = 0.f;
  #pragma unroll
  for (int j = 0; j < 8; ++j) {
    int col = j * 16 + lc;
    v16h bf;
    #pragma unroll
    for (int e = 0; e < 16; ++e) bf[e] = w1h[j * 512 + kmap(half, e) * 16 + lc];
    v8f c = {0.f, 0.f, 0.f, 0.f, 0.f, 0.f, 0.f, 0.f};
    c = __builtin_amdgcn_wmma_f32_16x16x32_f16(false, a, false, bf,
                                               (short)0, c, false, false);
    float bb = b1[col], ww = w2[col];
    #pragma unroll
    for (int r = 0; r < 8; ++r) acc[r] += gelu_f(c[r] + bb) * ww;
  }
  #pragma unroll
  for (int m = 1; m < 16; m <<= 1)
    #pragma unroll
    for (int r = 0; r < 8; ++r) acc[r] += __shfl_xor(acc[r], m, 32);
  if (lc == 0) {
    float bias = b2[0];
    #pragma unroll
    for (int r = 0; r < 8; ++r)
      out[tile * 16 + half * 8 + r] = acc[r] + bias;
  }
}

// ---------------------------------------------------------------------------
extern "C" void kernel_launch(void* const* d_in, const int* in_sizes, int n_in,
                              void* d_out, int out_size, void* d_ws, size_t ws_size,
                              hipStream_t stream) {
  (void)in_sizes; (void)n_in; (void)out_size; (void)ws_size;
  const float* x_in    = (const float*)d_in[0];
  const float* fc0_w   = (const float*)d_in[1];
  const float* fc0_b   = (const float*)d_in[2];
  const float* spec_wA = (const float*)d_in[3];
  const float* spec_wB = (const float*)d_in[4];
  const float* conv_w  = (const float*)d_in[5];
  const float* conv_b  = (const float*)d_in[6];
  const float* gate_w  = (const float*)d_in[7];
  const float* gate_b  = (const float*)d_in[8];
  const float* fc1_w   = (const float*)d_in[9];
  const float* fc1_b   = (const float*)d_in[10];
  const float* fc2_w   = (const float*)d_in[11];
  const float* fc2_b   = (const float*)d_in[12];
  float* out = (float*)d_out;

  float* ws   = (float*)d_ws;
  float* h    = ws;                       // 30,146,560
  float* Hz   = h   + 30146560;           // 10,485,760
  float* Hyz  = Hz  + 10485760;           //  2,621,440
  float* Hc   = Hyz + 2621440;            //    655,360
  float* Gc   = Hc  + 655360;             //    655,360
  float* Gx   = Gc  + 655360;             //  2,621,440
  float* Gy   = Gx  + 2621440;            // 10,485,760
  float* Gate = Gy  + 10485760;           //        256
  float* Trig = Gate + 256;               //        736
  _Float16* Bas = (_Float16*)(Trig + 736);
  _Float16* BZF = Bas;
  _Float16* BYF = Bas + 1024;
  _Float16* BXI = Bas + 5120;
  _Float16* W1H = Bas + 9216;             //  4096 halfs

  init_basis<<<1, 256, 0, stream>>>(Bas, Trig, fc1_w);
  fc0_kernel<<<5888, 256, 0, stream>>>(x_in, fc0_w, fc0_b, h);

  for (int l = 0; l < 4; ++l) {
    dft_gemm<0><<<dim3(40960, 1), 32, 0, stream>>>(h,   BZF, Hz);
    dft_gemm<1><<<dim3(5120,  2), 32, 0, stream>>>(Hz,  BYF, Hyz);
    dft_gemm<2><<<dim3(1280,  2), 32, 0, stream>>>(Hyz, BYF, Hc);
    gate_kernel<<<8, 256, 0, stream>>>(Hc, gate_w, gate_b, Gate, l);
    spectral_kernel<<<1280, 256, 0, stream>>>(Hc, Gate, spec_wA, spec_wB, Gc, l);
    dft_gemm<3><<<dim3(1280,  8), 32, 0, stream>>>(Gc,  BXI, Gx);
    dft_gemm<4><<<dim3(5120,  8), 32, 0, stream>>>(Gx,  BXI, Gy);
    zi_conv_kernel<<<32768, 128, 0, stream>>>(Gy, h, conv_w, conv_b, Trig, l);
  }

  fc12_kernel<<<81920, 32, 0, stream>>>(h, W1H, fc1_b, fc2_w, fc2_b, out);
}
